// TriangleAttention_72627896975618
// MI455X (gfx1250) — matrix operand order repaired and checked by reading, not compiled, not run on hardware
//
#include <hip/hip_runtime.h>
#include <hip/hip_bf16.h>

// TriangleAttention (starting node) for MI455X / gfx1250, wave32.
// Transposed dataflow everywhere (lane owns one row, 8 contiguous cols).
// Kernel 1/3: 256 rows per WG with register-resident X fragments so weight
// staging is amortized 4x.  Kernel 2: TDM-staged K tile, shared-B AV loop,
// bf16 triangle-bias plane.

#define SEQ    256
#define CDIM   128
#define NHEADS 4
#define DHID   32
#define INFB   1.0e9f
#define LNEPS  1e-5f

typedef __attribute__((ext_vector_type(16))) __bf16 v16bf;
typedef __attribute__((ext_vector_type(8)))  float  v8f;
typedef __attribute__((ext_vector_type(4)))  unsigned int u32x4;
typedef __attribute__((ext_vector_type(2)))  unsigned int u32x2;
typedef __attribute__((ext_vector_type(8)))  int    i32x8;
typedef __attribute__((ext_vector_type(4)))  int    i32x4;

union Frag16 { v16bf v; u32x4 q[2]; };
union Pack8  { u32x4 u; __bf16 b[8]; };
union Pack4  { u32x2 u; __bf16 b[4]; };

__device__ __forceinline__ v8f vzero8() {
  v8f z = {0.f,0.f,0.f,0.f,0.f,0.f,0.f,0.f};
  return z;
}

__device__ __forceinline__ v8f wmma_bf16(const Frag16& a, const Frag16& b, v8f c) {
  return __builtin_amdgcn_wmma_f32_16x16x32_bf16(false, a.v, false, b.v,
                                                 (short)0, c, false, false);
}

// A-fragment (16x32 bf16): lanes 0-15 K=kbase+0..15; lanes 16-31 +16.
__device__ __forceinline__ Frag16 ld_afrag(const unsigned char* base, int row,
                                           int strideB, int kbase, int lane) {
  int ks = kbase + ((lane & 16) ? 16 : 0);
  const unsigned char* p = base + row * strideB + 2 * ks;
  Frag16 f;
  f.q[0] = *(const u32x4*)(p);
  f.q[1] = *(const u32x4*)(p + 16);
  return f;
}

// B-fragment (32x16 bf16): row index = N; lanes 0-15 K=kbase+0..15, 16-31 +16.
__device__ __forceinline__ Frag16 ld_bfrag(const unsigned char* base, int row,
                                           int strideB, int kbase, int lane) {
  int ks = kbase + ((lane & 16) ? 16 : 0);
  const unsigned char* p = base + row * strideB + 2 * ks;
  Frag16 f;
  f.q[0] = *(const u32x4*)(p);
  f.q[1] = *(const u32x4*)(p + 16);
  return f;
}

// ---------------------------------------------------------------------------
// Kernel 1: LayerNorm + Q/K/V/G/tri projections.  256 rows per WG, 128 thr.
// OUT^T = W^T * X^T : A = wT tiles, B = register-resident xn fragments.
// ---------------------------------------------------------------------------
__device__ __forceinline__ void stage_w128(unsigned char* wT,
                                           const float* __restrict__ W, int tid) {
  for (int e = tid; e < 16384; e += 128) {
    int k = e >> 7, n = e & 127;
    ((__bf16*)(wT + n * 272))[k] = (__bf16)W[e];
  }
}

__device__ __forceinline__ void stage_tri(unsigned char* wT,
                                          const float* __restrict__ W, int tid) {
  for (int e = tid; e < 2048; e += 128) {
    int k = e >> 4, n = e & 15;
    float val = (n < 4) ? W[k * 4 + n] : 0.0f;
    ((__bf16*)(wT + n * 272))[k] = (__bf16)val;
  }
}

template <int WSEL>
__device__ __forceinline__ void proj_gemm(
    const Frag16 (&bfrag)[4], const unsigned char* wT,
    int rowbase, int wave, int lane, const float* __restrict__ g_b,
    __bf16* __restrict__ Qg, __bf16* __restrict__ Kg, __bf16* __restrict__ Vg,
    __bf16* __restrict__ Gg, __bf16* __restrict__ TriB)
{
  const int nl   = lane & 15;
  const int madd = (lane & 16) ? 8 : 0;
  const int row  = rowbase + wave * 16 + nl;   // this lane's fixed output row
  const int i = row >> 8, j = row & 255;
  const int mtiles = (WSEL == 4) ? 1 : 8;
  #pragma unroll
  for (int mt = 0; mt < mtiles; ++mt) {
    Frag16 a0 = ld_afrag(wT, mt * 16 + nl, 272, 0,  lane);
    Frag16 a1 = ld_afrag(wT, mt * 16 + nl, 272, 32, lane);
    Frag16 a2 = ld_afrag(wT, mt * 16 + nl, 272, 64, lane);
    Frag16 a3 = ld_afrag(wT, mt * 16 + nl, 272, 96, lane);
    v8f acc = wmma_bf16(a0, bfrag[0], vzero8());
    acc = wmma_bf16(a1, bfrag[1], acc);
    acc = wmma_bf16(a2, bfrag[2], acc);
    acc = wmma_bf16(a3, bfrag[3], acc);

    const int col0 = mt * 16 + madd;           // 8 consecutive output columns
    if constexpr (WSEL <= 2) {
      Pack8 p;
      #pragma unroll
      for (int v = 0; v < 8; ++v)
        p.b[v] = (__bf16)(WSEL == 0 ? acc[v] * 0.17677669529663687f : acc[v]);
      __bf16* dst = (WSEL == 0) ? Qg : ((WSEL == 1) ? Kg : Vg);
      *(u32x4*)(dst + (((size_t)(i * NHEADS + (col0 >> 5))) * SEQ + j) * DHID +
                (col0 & 31)) = p.u;
    } else if constexpr (WSEL == 3) {
      float4 gb0 = *(const float4*)(g_b + col0);
      float4 gb1 = *(const float4*)(g_b + col0 + 4);
      float gb[8] = {gb0.x, gb0.y, gb0.z, gb0.w, gb1.x, gb1.y, gb1.z, gb1.w};
      Pack8 p;
      #pragma unroll
      for (int v = 0; v < 8; ++v) {
        float gx = acc[v] + gb[v];
        p.b[v] = (__bf16)__builtin_amdgcn_rcpf(1.0f + __expf(-gx)); // sigmoid
      }
      *(u32x4*)(Gg + (size_t)row * CDIM + col0) = p.u;
    } else {
      if (madd == 0) {                          // tri: h = v (only h < 4 valid)
        #pragma unroll
        for (int v = 0; v < 4; ++v)
          TriB[((size_t)(v * SEQ + i)) * SEQ + j] = (__bf16)acc[v];
      }
    }
  }
}

template <int WSEL>
__device__ __forceinline__ void proj_pass(
    const Frag16 (&bfrag)[4][4], unsigned char* wT, const float* __restrict__ W,
    int tid, int R0, int wave, int lane, const float* __restrict__ g_b,
    __bf16* __restrict__ Qg, __bf16* __restrict__ Kg, __bf16* __restrict__ Vg,
    __bf16* __restrict__ Gg, __bf16* __restrict__ TriB)
{
  __syncthreads();                     // protect wT against previous readers
  if constexpr (WSEL < 4) stage_w128(wT, W, tid);
  else                    stage_tri(wT, W, tid);
  __syncthreads();
  #pragma unroll
  for (int rb = 0; rb < 4; ++rb)
    proj_gemm<WSEL>(bfrag[rb], wT, R0 + rb * 64, wave, lane, g_b,
                    Qg, Kg, Vg, Gg, TriB);
}

__global__ void __launch_bounds__(128) k_ln_proj(
    const float* __restrict__ x,    const float* __restrict__ ln_w,
    const float* __restrict__ ln_b, const float* __restrict__ tri_w,
    const float* __restrict__ q_w,  const float* __restrict__ k_w,
    const float* __restrict__ v_w,  const float* __restrict__ g_w,
    const float* __restrict__ g_b,
    __bf16* __restrict__ Qg, __bf16* __restrict__ Kg, __bf16* __restrict__ Vg,
    __bf16* __restrict__ Gg, __bf16* __restrict__ TriB)
{
  __shared__ __align__(16) unsigned char lds[17408 + 34816];
  unsigned char* xn = lds;            // 64 rows x 272B, reused per row-block
  unsigned char* wT = lds + 17408;    // 128 n-rows x 272B (k contiguous)

  const int tid  = threadIdx.x;
  const int lane = tid & 31;
  const int wave = tid >> 5;
  const int R0   = blockIdx.x * 256;

  // LayerNorm 4 row-blocks; each wave touches only its own 16 LDS rows, so
  // the LN-store -> bfrag-load path is same-wave (no barriers needed).
  const int c0 = lane * 4;
  float4 lw = *(const float4*)(ln_w + c0);
  float4 lb = *(const float4*)(ln_b + c0);
  Frag16 bfrag[4][4];                  // X^T fragments, register resident
  #pragma unroll
  for (int rb = 0; rb < 4; ++rb) {
    for (int r = 0; r < 16; ++r) {
      int rloc = wave * 16 + r;
      int rg   = R0 + rb * 64 + rloc;
      float4 xv = *(const float4*)(x + (size_t)rg * CDIM + c0);
      float s = xv.x + xv.y + xv.z + xv.w;
      #pragma unroll
      for (int m = 16; m >= 1; m >>= 1) s += __shfl_xor(s, m, 32);
      float mu = s * (1.0f / 128.0f);
      float dx = xv.x - mu, dy = xv.y - mu, dz = xv.z - mu, dw = xv.w - mu;
      float vs = dx*dx + dy*dy + dz*dz + dw*dw;
      #pragma unroll
      for (int m = 16; m >= 1; m >>= 1) vs += __shfl_xor(vs, m, 32);
      float rstd = rsqrtf(vs * (1.0f / 128.0f) + LNEPS);
      Pack4 p;
      p.b[0] = (__bf16)(dx * rstd * lw.x + lb.x);
      p.b[1] = (__bf16)(dy * rstd * lw.y + lb.y);
      p.b[2] = (__bf16)(dz * rstd * lw.z + lb.z);
      p.b[3] = (__bf16)(dw * rstd * lw.w + lb.w);
      *(u32x2*)(xn + rloc * 272 + c0 * 2) = p.u;     // one b64 store
    }
    #pragma unroll
    for (int kb = 0; kb < 4; ++kb)
      bfrag[rb][kb] = ld_bfrag(xn, wave * 16 + (lane & 15), 272, kb * 32, lane);
  }

  proj_pass<0>(bfrag, wT, q_w,  tid, R0, wave, lane, g_b, Qg, Kg, Vg, Gg, TriB);
  proj_pass<1>(bfrag, wT, k_w,  tid, R0, wave, lane, g_b, Qg, Kg, Vg, Gg, TriB);
  proj_pass<2>(bfrag, wT, v_w,  tid, R0, wave, lane, g_b, Qg, Kg, Vg, Gg, TriB);
  proj_pass<3>(bfrag, wT, g_w,  tid, R0, wave, lane, g_b, Qg, Kg, Vg, Gg, TriB);
  proj_pass<4>(bfrag, wT, tri_w,tid, R0, wave, lane, g_b, Qg, Kg, Vg, Gg, TriB);
}

// ---------------------------------------------------------------------------
// Kernel 2: attention for one (i,h).  128 threads, 4 waves x 4 q-tiles.
// L^T = K*Q^T (TDM-staged K), per-lane softmax (one shuffle), P^T staged with
// packed b128 writes, O^T = V^T * P^T with shared B fragments.
// ---------------------------------------------------------------------------
__global__ void __launch_bounds__(128) k_attn(
    const __bf16* __restrict__ Qg, const __bf16* __restrict__ Kg,
    const __bf16* __restrict__ Vg, const float* __restrict__ mask,
    const __bf16* __restrict__ TriB, __bf16* __restrict__ Og)
{
  __shared__ __align__(16) unsigned char lds[20480 + 8704 + 1024 + 33792];
  unsigned char* Ks    = lds;                       // 256 x 80B  [j][d] (+pad)
  unsigned char* VT    = lds + 20480;               // 32 x 272B  [d][j]
  float*         maskb = (float*)(lds + 29184);     // 256 f32
  unsigned char* stage = lds + 30208;               // 4 waves x 16 x 528B

  const int tid  = threadIdx.x;
  const int lane = tid & 31;
  const int wave = tid >> 5;
  const int i    = blockIdx.x >> 2;
  const int h    = blockIdx.x & 3;
  const size_t base = (size_t)(i * NHEADS + h) * (SEQ * DHID);

  // --- TDM: K tile [256 x 64B] global -> LDS rows padded to 80B. wave0 only,
  // scalarized branch so the EXEC-ignoring tensor op issues exactly once.
  if ((__builtin_amdgcn_readfirstlane((int)threadIdx.x) >> 5) == 0) {
    unsigned long long ga = (unsigned long long)(const void*)(Kg + base);
    unsigned int ldsoff = (unsigned int)(unsigned long long)(const void*)Ks;
    u32x4 g0;
    g0[0] = 1u;                                        // count=1, no gather
    g0[1] = ldsoff;                                    // lds_addr
    g0[2] = (unsigned int)(ga & 0xFFFFFFFFu);          // global_addr lo
    g0[3] = (unsigned int)((ga >> 32) & 0x1FFFFFFu) | (2u << 30); // hi | type=2
    i32x8 g1;
    g1[0] = (int)((3u << 16) |   // data_size: 8B units
                  (1u << 20) |   // pad_enable
                  (3u << 22) |   // pad_interval: 16 DWORDs (64B)
                  (3u << 25));   // pad_amount: 4 DWORDs (16B)
    g1[1] = (int)(8u << 16);     // tensor_dim0 = 8   (64B row / 8B)
    g1[2] = (int)(256u << 16);   // tensor_dim1 = 256
    g1[3] = (int)(8u << 16);     // tile_dim0 = 8
    g1[4] = (int)(256u);         // tile_dim1 = 256, tile_dim2 = 0
    g1[5] = (int)(8u);           // tensor_dim0_stride = 8
    g1[6] = 0;
    g1[7] = 0;
    i32x4 gz = {0, 0, 0, 0};
#if __clang_major__ >= 23
    i32x8 gz8 = {0, 0, 0, 0, 0, 0, 0, 0};
    __builtin_amdgcn_tensor_load_to_lds(g0, g1, gz, gz, gz8, 0);
#else
    __builtin_amdgcn_tensor_load_to_lds(g0, g1, gz, gz, 0);
#endif
    __builtin_amdgcn_s_wait_tensorcnt(0);
  }

  // manual staging of V (transposed) and mask bias in parallel with the TDM
  for (int e = tid; e < SEQ * DHID; e += 128) {
    int j = e >> 5, d = e & 31;
    ((__bf16*)(VT + d * 272))[j] = Vg[base + e];
  }
  for (int e = tid; e < SEQ; e += 128)
    maskb[e] = INFB * (mask[i * SEQ + e] - 1.0f);
  __syncthreads();

  unsigned char* mystage = stage + wave * (16 * 528);
  const int nl   = lane & 15;
  const int madd = (lane & 16) ? 8 : 0;

  for (int t = 0; t < 4; ++t) {
    const int q0 = (wave * 4 + t) * 16;
    const int q  = q0 + nl;                 // this lane's fixed query row

    // init acc with (mask + tri) bias, transposed layout (key = nt*16+madd+v)
    v8f acc[16];
    const __bf16* tb = TriB + ((size_t)(h * SEQ + q)) * SEQ;
    #pragma unroll
    for (int nt = 0; nt < 16; ++nt) {
      int k0 = nt * 16 + madd;
      Pack8 tp;
      tp.u = *(const u32x4*)(tb + k0);
      float4 m0 = *(const float4*)(maskb + k0);
      float4 m1 = *(const float4*)(maskb + k0 + 4);
      acc[nt][0] = (float)tp.b[0] + m0.x; acc[nt][1] = (float)tp.b[1] + m0.y;
      acc[nt][2] = (float)tp.b[2] + m0.z; acc[nt][3] = (float)tp.b[3] + m0.w;
      acc[nt][4] = (float)tp.b[4] + m1.x; acc[nt][5] = (float)tp.b[5] + m1.y;
      acc[nt][6] = (float)tp.b[6] + m1.z; acc[nt][7] = (float)tp.b[7] + m1.w;
    }

    // Q^T B-fragment straight from global (row = q, contiguous d run)
    Frag16 bq;
    {
      const __bf16* p = Qg + base + q * DHID + ((lane & 16) ? 16 : 0);
      bq.q[0] = *(const u32x4*)(p);
      bq.q[1] = *(const u32x4*)(p + 8);
    }

    // L^T = K * Q^T: 16 key tiles, A-fragment double buffered
    Frag16 ak = ld_afrag(Ks, nl, 80, 0, lane);
    #pragma unroll
    for (int nt = 0; nt < 16; ++nt) {
      Frag16 an = ld_afrag(Ks, ((nt + 1) & 15) * 16 + nl, 80, 0, lane);
      acc[nt] = wmma_bf16(ak, bq, acc[nt]);
      ak = an;
    }

    // softmax over keys: fully in-register + ONE cross-lane exchange (l^16)
    float m = -3.0e38f;
    #pragma unroll
    for (int nt = 0; nt < 16; ++nt)
      #pragma unroll
      for (int v = 0; v < 8; ++v) m = fmaxf(m, acc[nt][v]);
    m = fmaxf(m, __shfl_xor(m, 16, 32));
    float s = 0.f;
    #pragma unroll
    for (int nt = 0; nt < 16; ++nt)
      #pragma unroll
      for (int v = 0; v < 8; ++v) {
        float e = __expf(acc[nt][v] - m);
        acc[nt][v] = e;
        s += e;
      }
    s += __shfl_xor(s, 16, 32);
    float inv = __builtin_amdgcn_rcpf(s);

    // stage P^T as [q][k] rows: 8 contiguous bf16 per nt -> packed b128 writes
    #pragma unroll
    for (int nt = 0; nt < 16; ++nt) {
      Pack8 p;
      #pragma unroll
      for (int v = 0; v < 8; ++v) p.b[v] = (__bf16)(acc[nt][v] * inv);
      *(u32x4*)(mystage + nl * 528 + 2 * (nt * 16 + madd)) = p.u;
    }

    // O^T = V^T * P^T: both d-tiles share each B fragment (one load, 2 wmma)
    v8f ao0 = vzero8(), ao1 = vzero8();
    #pragma unroll
    for (int kb = 0; kb < 8; ++kb) {
      Frag16 bp  = ld_bfrag(mystage, nl, 528, kb * 32, lane);
      Frag16 av0 = ld_afrag(VT, nl,      272, kb * 32, lane);
      Frag16 av1 = ld_afrag(VT, 16 + nl, 272, kb * 32, lane);
      ao0 = wmma_bf16(av0, bp, ao0);
      ao1 = wmma_bf16(av1, bp, ao1);
    }
    __bf16* orow = Og + ((size_t)(i * SEQ + q)) * CDIM + h * DHID + madd;
    Pack8 p0, p1;
    #pragma unroll
    for (int v = 0; v < 8; ++v) {
      p0.b[v] = (__bf16)ao0[v];
      p1.b[v] = (__bf16)ao1[v];
    }
    *(u32x4*)(orow)      = p0.u;
    *(u32x4*)(orow + 16) = p1.u;
  }
}

// ---------------------------------------------------------------------------
// Kernel 3: out = (O .* G) @ o_w + o_b, computed as OUT^T = o_w^T * act^T.
// 256 rows per WG, 128 threads; register-resident activation fragments.
// ---------------------------------------------------------------------------
__global__ void __launch_bounds__(128) k_out(
    const __bf16* __restrict__ Og, const __bf16* __restrict__ Gg,
    const float* __restrict__ o_w, const float* __restrict__ o_b,
    float* __restrict__ out)
{
  __shared__ __align__(16) unsigned char lds[17408 + 34816];
  unsigned char* A  = lds;            // 64 x 272B bf16, reused per row-block
  unsigned char* wT = lds + 17408;    // 128 n-rows x 272B

  const int tid  = threadIdx.x;
  const int lane = tid & 31;
  const int wave = tid >> 5;
  const int R0   = blockIdx.x * 256;

  stage_w128(wT, o_w, tid);

  const int nl   = lane & 15;
  const int madd = (lane & 16) ? 8 : 0;

  Frag16 bf[4][4];                    // gated-activation fragments (4 blocks)
  #pragma unroll
  for (int rb = 0; rb < 4; ++rb) {
    __syncthreads();                  // protect A against previous readers
    for (int e = tid; e < 8192; e += 128) {
      int r = e >> 7, c = e & 127;
      size_t idx = (size_t)(R0 + rb * 64 + r) * CDIM + c;
      float a = (float)Og[idx] * (float)Gg[idx];
      ((__bf16*)(A + r * 272))[c] = (__bf16)a;
    }
    __syncthreads();
    #pragma unroll
    for (int kb = 0; kb < 4; ++kb)
      bf[rb][kb] = ld_bfrag(A, wave * 16 + nl, 272, kb * 32, lane);
  }
  __syncthreads();                    // wT + all fragments ready

  #pragma unroll
  for (int rb = 0; rb < 4; ++rb) {
    const int row = R0 + rb * 64 + wave * 16 + nl;
    #pragma unroll
    for (int mt = 0; mt < 8; ++mt) {
      Frag16 a0 = ld_afrag(wT, mt * 16 + nl, 272, 0,  lane);
      Frag16 a1 = ld_afrag(wT, mt * 16 + nl, 272, 32, lane);
      Frag16 a2 = ld_afrag(wT, mt * 16 + nl, 272, 64, lane);
      Frag16 a3 = ld_afrag(wT, mt * 16 + nl, 272, 96, lane);
      v8f acc = wmma_bf16(a0, bf[rb][0], vzero8());
      acc = wmma_bf16(a1, bf[rb][1], acc);
      acc = wmma_bf16(a2, bf[rb][2], acc);
      acc = wmma_bf16(a3, bf[rb][3], acc);

      const int col0 = mt * 16 + madd;
      float4 ob0 = *(const float4*)(o_b + col0);
      float4 ob1 = *(const float4*)(o_b + col0 + 4);
      float4 r0 = make_float4(acc[0] + ob0.x, acc[1] + ob0.y,
                              acc[2] + ob0.z, acc[3] + ob0.w);
      float4 r1 = make_float4(acc[4] + ob1.x, acc[5] + ob1.y,
                              acc[6] + ob1.z, acc[7] + ob1.w);
      *(float4*)(out + (size_t)row * CDIM + col0)     = r0;
      *(float4*)(out + (size_t)row * CDIM + col0 + 4) = r1;
    }
  }
}

// ---------------------------------------------------------------------------
extern "C" void kernel_launch(void* const* d_in, const int* in_sizes, int n_in,
                              void* d_out, int out_size, void* d_ws, size_t ws_size,
                              hipStream_t stream) {
  const float* x     = (const float*)d_in[0];
  const float* mask  = (const float*)d_in[1];
  const float* ln_w  = (const float*)d_in[2];
  const float* ln_b  = (const float*)d_in[3];
  const float* tri_w = (const float*)d_in[4];
  const float* q_w   = (const float*)d_in[5];
  const float* k_w   = (const float*)d_in[6];
  const float* v_w   = (const float*)d_in[7];
  const float* g_w   = (const float*)d_in[8];
  const float* g_b   = (const float*)d_in[9];
  const float* o_w   = (const float*)d_in[10];
  const float* o_b   = (const float*)d_in[11];
  float* out = (float*)d_out;

  // workspace: 5 x 16MB bf16 tensors + 512KB bf16 TriB  (~81 MB)
  unsigned char* ws = (unsigned char*)d_ws;
  const size_t SZ = (size_t)SEQ * SEQ * CDIM * 2;   // 16,777,216 B
  __bf16* Qg   = (__bf16*)(ws);
  __bf16* Kg   = (__bf16*)(ws + SZ);
  __bf16* Vg   = (__bf16*)(ws + 2 * SZ);
  __bf16* Gg   = (__bf16*)(ws + 3 * SZ);
  __bf16* Og   = (__bf16*)(ws + 4 * SZ);
  __bf16* TriB = (__bf16*)(ws + 5 * SZ);

  k_ln_proj<<<256, 128, 0, stream>>>(x, ln_w, ln_b, tri_w, q_w, k_w, v_w,
                                     g_w, g_b, Qg, Kg, Vg, Gg, TriB);
  k_attn<<<1024, 128, 0, stream>>>(Qg, Kg, Vg, mask, TriB, Og);
  k_out<<<256, 128, 0, stream>>>(Og, Gg, o_w, o_b, out);
}
